// EnhancedGraphSAGE_69526930588461
// MI455X (gfx1250) — matrix operand order repaired and checked by reading, not compiled, not run on hardware
//
#include <hip/hip_runtime.h>
#include <hip/hip_bf16.h>
#include <math.h>

typedef _Float16 v16h __attribute__((ext_vector_type(16)));
typedef float    v8f  __attribute__((ext_vector_type(8)));

#define H_FEATS 64
#define HEADS 4
#define GATF (HEADS * H_FEATS)   // 256

// ---------------------------------------------------------------------------
// Weight packing: fp32 B[K,NcolsReal] row-major  ->  f16 WMMA-B fragments.
// Layout: Bp[((ct*nK + ks)*32 + lane)*16 + i] = B[k, n]  (0 if n OOB)
//   k = ks*32 + (lane>>4)*16 + i,  n = ct*16 + (lane&15)
// Each lane then loads its 16 halves as one contiguous 32B chunk.
// ---------------------------------------------------------------------------
__global__ void pack_b_kernel(const float* __restrict__ B, _Float16* __restrict__ Bp,
                              int K, int NcolsReal, int NcolsPad) {
  int idx = blockIdx.x * blockDim.x + threadIdx.x;
  int nK = K >> 5;
  int total = (NcolsPad >> 4) * nK * 512;
  if (idx >= total) return;
  int i    = idx & 15;
  int lane = (idx >> 4) & 31;
  int blk  = idx >> 9;                 // ct*nK + ks
  int ct   = blk / nK;
  int ks   = blk - ct * nK;
  int k = ks * 32 + (lane >> 4) * 16 + i;
  int n = ct * 16 + (lane & 15);
  Bp[idx] = (n < NcolsReal) ? (_Float16)B[(size_t)k * NcolsReal + n] : (_Float16)0.0f;
}

// ---------------------------------------------------------------------------
// WMMA GEMM:  C = act( A[M,K]@B + A2[M,K2]@B2 + bias ),  B pre-packed f16.
// blockDim = (32, 4): each wave owns one 16-col tile; one 16-row tile per block.x.
// ---------------------------------------------------------------------------
__device__ __forceinline__ v8f wmma_kloop(const float* __restrict__ A,
                                          const _Float16* __restrict__ Bp,
                                          int rowBase, int ct, int K,
                                          int lane, v8f acc) {
  const int m  = lane & 15;
  const int kg = lane >> 4;
  const int nK = K >> 5;
  const float* __restrict__ arow = A + (size_t)(rowBase + m) * (size_t)K;
  const _Float16* __restrict__ bptr = Bp + ((size_t)ct * nK * 512) + lane * 16;
  for (int ks = 0; ks < nK; ++ks) {
    const int k0 = ks * 32;
    v16h af;
#pragma unroll
    for (int i = 0; i < 8; ++i) {
      af[i]     = (_Float16)arow[k0 + kg * 16 + i];
      af[i + 8] = (_Float16)arow[k0 + kg * 16 + 8 + i];
    }
    v16h bf = *(const v16h*)bptr;
    bptr += 512;
    acc = __builtin_amdgcn_wmma_f32_16x16x32_f16(
        false, af, false, bf, (short)0, acc, false, false);
  }
  return acc;
}

__global__ void gemm_wmma_kernel(const float* __restrict__ A,
                                 const _Float16* __restrict__ Bp,
                                 const float* __restrict__ A2,
                                 const _Float16* __restrict__ Bp2,
                                 const float* __restrict__ bias,
                                 float* __restrict__ C,
                                 int K, int K2, int NcolsReal, int NcolsPad,
                                 int doRelu) {
  const int lane = threadIdx.x;                       // 0..31
  const int ct   = blockIdx.y * blockDim.y + threadIdx.y;
  if (ct >= (NcolsPad >> 4)) return;                  // uniform per wave
  const int rowBase = blockIdx.x * 16;
  const int kg = lane >> 4;
  const int n  = ct * 16 + (lane & 15);
  const bool nok = (n < NcolsReal);

  v8f acc = {0.f, 0.f, 0.f, 0.f, 0.f, 0.f, 0.f, 0.f};
  acc = wmma_kloop(A, Bp, rowBase, ct, K, lane, acc);
  if (K2 > 0)
    acc = wmma_kloop(A2, Bp2, rowBase, ct, K2, lane, acc);

  const float bv = (bias != nullptr && nok) ? bias[n] : 0.0f;
#pragma unroll
  for (int r = 0; r < 8; ++r) {
    float v = acc[r] + bv;
    if (doRelu) v = fmaxf(v, 0.0f);
    if (nok) C[(size_t)(rowBase + r + 8 * kg) * (size_t)NcolsReal + n] = v;
  }
}

// ---------------------------------------------------------------------------
// Elementwise / graph kernels (L2-resident gather/scatter; atomic-bound)
// ---------------------------------------------------------------------------
__global__ void fill_kernel(float* __restrict__ p, float v, int n) {
  int i = blockIdx.x * blockDim.x + threadIdx.x;
  if (i < n) p[i] = v;
}

__global__ void degree_kernel(const int* __restrict__ dst,
                              float* __restrict__ deg, int E) {
  int e = blockIdx.x * blockDim.x + threadIdx.x;
  if (e < E) atomicAdd(&deg[dst[e]], 1.0f);
}

// wave per edge, 64 features (2 per lane)
__global__ void sage_scatter_kernel(const float* __restrict__ h,
                                    const int* __restrict__ src,
                                    const int* __restrict__ dst,
                                    float* __restrict__ s, int E) {
  int gid  = blockIdx.x * blockDim.x + threadIdx.x;
  int e    = gid >> 5;
  int lane = gid & 31;
  if (e >= E) return;
  size_t u = (size_t)src[e] * H_FEATS;
  size_t v = (size_t)dst[e] * H_FEATS;
  atomicAdd(&s[v + lane],      h[u + lane]);
  atomicAdd(&s[v + 32 + lane], h[u + 32 + lane]);
}

__global__ void neigh_div_kernel(float* __restrict__ s,
                                 const float* __restrict__ deg, int total) {
  int i = blockIdx.x * blockDim.x + threadIdx.x;
  if (i < total) s[i] = s[i] / fmaxf(deg[i >> 6], 1.0f);
}

// el[n*4+hd], er[n*4+hd] from feat [N,256]
__global__ void attn_proj_kernel(const float* __restrict__ feat,
                                 const float* __restrict__ al,
                                 const float* __restrict__ ar,
                                 float* __restrict__ el,
                                 float* __restrict__ er, int N) {
  int i = blockIdx.x * blockDim.x + threadIdx.x;
  if (i >= N * HEADS) return;
  int hd = i & 3;
  int nn = i >> 2;
  const float* f = feat + (size_t)nn * GATF + hd * H_FEATS;
  const float* a = al + hd * H_FEATS;
  const float* b = ar + hd * H_FEATS;
  float sl = 0.f, sr = 0.f;
  for (int j = 0; j < H_FEATS; ++j) {
    float fv = f[j];
    sl += fv * a[j];
    sr += fv * b[j];
  }
  el[i] = sl;
  er[i] = sr;
}

__device__ __forceinline__ float leaky02(float x) {
  return x >= 0.0f ? x : 0.2f * x;
}

// float atomic-max via sign-aware int/uint punning (valid with -inf init)
__device__ __forceinline__ void atomicMaxF(float* addr, float v) {
  if (v >= 0.0f) atomicMax((int*)addr, __float_as_int(v));
  else           atomicMin((unsigned int*)addr, __float_as_uint(v));
}

__global__ void edge_max_kernel(const int* __restrict__ src,
                                const int* __restrict__ dst,
                                const float* __restrict__ el,
                                const float* __restrict__ er,
                                float* __restrict__ m, int E) {
  int e = blockIdx.x * blockDim.x + threadIdx.x;
  if (e >= E) return;
  int u = src[e] * HEADS, v = dst[e] * HEADS;
#pragma unroll
  for (int hd = 0; hd < HEADS; ++hd)
    atomicMaxF(&m[v + hd], leaky02(el[u + hd] + er[v + hd]));
}

__global__ void edge_expsum_kernel(const int* __restrict__ src,
                                   const int* __restrict__ dst,
                                   const float* __restrict__ el,
                                   const float* __restrict__ er,
                                   const float* __restrict__ m,
                                   float* __restrict__ z, int E) {
  int e = blockIdx.x * blockDim.x + threadIdx.x;
  if (e >= E) return;
  int u = src[e] * HEADS, v = dst[e] * HEADS;
#pragma unroll
  for (int hd = 0; hd < HEADS; ++hd) {
    float x = leaky02(el[u + hd] + er[v + hd]);
    atomicAdd(&z[v + hd], __expf(x - m[v + hd]));
  }
}

// wave per edge; lane covers 8 contiguous feats; head = lane>>3
__global__ void gat_scatter_kernel(const int* __restrict__ src,
                                   const int* __restrict__ dst,
                                   const float* __restrict__ feat,
                                   const float* __restrict__ el,
                                   const float* __restrict__ er,
                                   const float* __restrict__ m,
                                   const float* __restrict__ z,
                                   float* __restrict__ gat, int E) {
  int gid  = blockIdx.x * blockDim.x + threadIdx.x;
  int e    = gid >> 5;
  int lane = gid & 31;
  if (e >= E) return;
  int u = src[e], v = dst[e];
  int hd = lane >> 3;
  float x = leaky02(el[u * HEADS + hd] + er[v * HEADS + hd]);
  float alpha = __expf(x - m[v * HEADS + hd]) / z[v * HEADS + hd];
  const float* fu = feat + (size_t)u * GATF + lane * 8;
  float* gv = gat + (size_t)v * GATF + lane * 8;
#pragma unroll
  for (int j = 0; j < 8; ++j) atomicAdd(&gv[j], alpha * fu[j]);
}

__global__ void add_bias256_kernel(float* __restrict__ gat,
                                   const float* __restrict__ bg, int total) {
  int i = blockIdx.x * blockDim.x + threadIdx.x;
  if (i < total) gat[i] += bg[i & (GATF - 1)];
}

// ---------------------------------------------------------------------------
// Host orchestration (graph-capture safe: kernel launches only)
// ---------------------------------------------------------------------------
extern "C" void kernel_launch(void* const* d_in, const int* in_sizes, int n_in,
                              void* d_out, int out_size, void* d_ws, size_t ws_size,
                              hipStream_t stream) {
  const float* x       = (const float*)d_in[0];
  const int*   src     = (const int*)d_in[1];
  const int*   dst     = (const int*)d_in[2];
  const float* W_embed = (const float*)d_in[3];
  const float* b_embed = (const float*)d_in[4];
  const float* Ws1     = (const float*)d_in[5];
  const float* Wn1     = (const float*)d_in[6];
  const float* bn1     = (const float*)d_in[7];
  const float* Ws2     = (const float*)d_in[8];
  const float* Wn2     = (const float*)d_in[9];
  const float* bn2     = (const float*)d_in[10];
  const float* Wg      = (const float*)d_in[11];
  const float* al      = (const float*)d_in[12];
  const float* ar      = (const float*)d_in[13];
  const float* bg      = (const float*)d_in[14];
  const float* W1      = (const float*)d_in[15];
  const float* b1      = (const float*)d_in[16];
  const float* W2      = (const float*)d_in[17];
  const float* b2      = (const float*)d_in[18];
  float* out = (float*)d_out;

  const int IN = 128, Hf = H_FEATS, C = 40, Cpad = 48;
  const int N = in_sizes[0] / IN;
  const int E = in_sizes[1];

  // ---- workspace layout: f16 packed weights first (128KB region), then fp32 ----
  _Float16* hp = (_Float16*)d_ws;
  _Float16* pWe  = hp;                 // 128x64  -> 8192
  _Float16* pWs1 = pWe  + 8192;        // 64x64   -> 4096
  _Float16* pWn1 = pWs1 + 4096;
  _Float16* pWs2 = pWn1 + 4096;
  _Float16* pWn2 = pWs2 + 4096;
  _Float16* pWg  = pWn2 + 4096;        // 64x256  -> 16384
  _Float16* pW1  = pWg  + 16384;       // 256x64  -> 16384
  _Float16* pW2  = pW1  + 16384;       // 64x48   -> 3072
  float* ws = (float*)((char*)d_ws + 131072);   // 128KB aligned offset

  float* h0   = ws;                      // [N,64]
  float* h1   = h0 + (size_t)N * Hf;     // [N,64]
  float* s    = h1 + (size_t)N * Hf;     // [N,64]
  float* feat = s  + (size_t)N * Hf;     // [N,256]
  float* gat  = feat + (size_t)N * GATF; // [N,256]
  float* deg  = gat  + (size_t)N * GATF; // [N]
  float* el   = deg + N;                 // [N,4]
  float* er   = el + (size_t)N * HEADS;  // [N,4]
  float* mx   = er + (size_t)N * HEADS;  // [N,4]
  float* z    = mx + (size_t)N * HEADS;  // [N,4]

  const int rowT = (N + 15) / 16;
  dim3 blkG(32, 4, 1);
  auto cdiv = [](int a, int b) { return (a + b - 1) / b; };
  auto gy = [&](int ncPad) { return cdiv(ncPad >> 4, 4); };

  // ---- pack weights to f16 WMMA fragments ----
  pack_b_kernel<<<cdiv(IN * Hf, 256), 256, 0, stream>>>(W_embed, pWe, IN, Hf, Hf);
  pack_b_kernel<<<cdiv(Hf * Hf, 256), 256, 0, stream>>>(Ws1, pWs1, Hf, Hf, Hf);
  pack_b_kernel<<<cdiv(Hf * Hf, 256), 256, 0, stream>>>(Wn1, pWn1, Hf, Hf, Hf);
  pack_b_kernel<<<cdiv(Hf * Hf, 256), 256, 0, stream>>>(Ws2, pWs2, Hf, Hf, Hf);
  pack_b_kernel<<<cdiv(Hf * Hf, 256), 256, 0, stream>>>(Wn2, pWn2, Hf, Hf, Hf);
  pack_b_kernel<<<cdiv(Hf * GATF, 256), 256, 0, stream>>>(Wg, pWg, Hf, GATF, GATF);
  pack_b_kernel<<<cdiv(GATF * Hf, 256), 256, 0, stream>>>(W1, pW1, GATF, Hf, Hf);
  pack_b_kernel<<<cdiv(Hf * Cpad, 256), 256, 0, stream>>>(W2, pW2, Hf, C, Cpad);

  // 1) h0 = x @ W_embed + b_embed
  gemm_wmma_kernel<<<dim3(rowT, gy(Hf)), blkG, 0, stream>>>(
      x, pWe, x, pWe, b_embed, h0, IN, 0, Hf, Hf, 0);

  // 2) in-degree
  fill_kernel<<<cdiv(N, 256), 256, 0, stream>>>(deg, 0.0f, N);
  degree_kernel<<<cdiv(E, 256), 256, 0, stream>>>(dst, deg, E);

  // 3) SAGE layer 1: h1 = relu(h0@Ws1 + mean_neigh(h0)@Wn1 + bn1)
  fill_kernel<<<cdiv(N * Hf, 256), 256, 0, stream>>>(s, 0.0f, N * Hf);
  sage_scatter_kernel<<<cdiv(E * 32, 256), 256, 0, stream>>>(h0, src, dst, s, E);
  neigh_div_kernel<<<cdiv(N * Hf, 256), 256, 0, stream>>>(s, deg, N * Hf);
  gemm_wmma_kernel<<<dim3(rowT, gy(Hf)), blkG, 0, stream>>>(
      h0, pWs1, s, pWn1, bn1, h1, Hf, Hf, Hf, Hf, 1);

  // 4) SAGE layer 2: h0 = relu(h1@Ws2 + mean_neigh(h1)@Wn2 + bn2)
  fill_kernel<<<cdiv(N * Hf, 256), 256, 0, stream>>>(s, 0.0f, N * Hf);
  sage_scatter_kernel<<<cdiv(E * 32, 256), 256, 0, stream>>>(h1, src, dst, s, E);
  neigh_div_kernel<<<cdiv(N * Hf, 256), 256, 0, stream>>>(s, deg, N * Hf);
  gemm_wmma_kernel<<<dim3(rowT, gy(Hf)), blkG, 0, stream>>>(
      h1, pWs2, s, pWn2, bn2, h0, Hf, Hf, Hf, Hf, 1);

  // 5) GAT projection: feat = h0 @ Wg   [N,256]
  gemm_wmma_kernel<<<dim3(rowT, gy(GATF)), blkG, 0, stream>>>(
      h0, pWg, h0, pWg, nullptr, feat, Hf, 0, GATF, GATF, 0);

  // 6) per-node attention logits
  attn_proj_kernel<<<cdiv(N * HEADS, 256), 256, 0, stream>>>(feat, al, ar, el, er, N);

  // 7) edge softmax: max, exp-sum
  fill_kernel<<<cdiv(N * HEADS, 256), 256, 0, stream>>>(mx, -INFINITY, N * HEADS);
  edge_max_kernel<<<cdiv(E, 256), 256, 0, stream>>>(src, dst, el, er, mx, E);
  fill_kernel<<<cdiv(N * HEADS, 256), 256, 0, stream>>>(z, 0.0f, N * HEADS);
  edge_expsum_kernel<<<cdiv(E, 256), 256, 0, stream>>>(src, dst, el, er, mx, z, E);

  // 8) weighted aggregation + head bias
  fill_kernel<<<cdiv(N * GATF, 256), 256, 0, stream>>>(gat, 0.0f, N * GATF);
  gat_scatter_kernel<<<cdiv(E * 32, 256), 256, 0, stream>>>(
      src, dst, feat, el, er, mx, z, gat, E);
  add_bias256_kernel<<<cdiv(N * GATF, 256), 256, 0, stream>>>(gat, bg, N * GATF);

  // 9) h1 = relu(gat @ W1 + b1)   [N,64]
  gemm_wmma_kernel<<<dim3(rowT, gy(Hf)), blkG, 0, stream>>>(
      gat, pW1, gat, pW1, b1, h1, GATF, 0, Hf, Hf, 1);

  // 10) out = h1 @ W2 + b2   [N,40] (padded to 48 cols, guarded stores)
  gemm_wmma_kernel<<<dim3(rowT, gy(Cpad)), blkG, 0, stream>>>(
      h1, pW2, h1, pW2, b2, out, Hf, 0, C, Cpad, 0);

  (void)n_in; (void)out_size; (void)ws_size;
}
